// attModel_5815385718858
// MI455X (gfx1250) — matrix operand order repaired and checked by reading, not compiled, not run on hardware
//
#include <hip/hip_runtime.h>
#include <hip/hip_bf16.h>

// ---------------------------------------------------------------------------
// MI455X (gfx1250) attention:
//  - all GEMMs via v_wmma_f32_16x16x32_bf16 (wave32)
//  - heavy GEMMs use LDS tiling (128x128 block / 256 threads), double-buffered
//    with GLOBAL_LOAD_ASYNC_TO_LDS_B128 (ASYNCcnt) when the toolchain has it.
// ---------------------------------------------------------------------------

typedef __bf16 bf16;
typedef __attribute__((ext_vector_type(16))) __bf16 v16bf;
typedef __attribute__((ext_vector_type(8)))  __bf16 v8bf;
typedef __attribute__((ext_vector_type(4)))  __bf16 v4bf;
typedef __attribute__((ext_vector_type(8)))  float  v8f;
typedef __attribute__((ext_vector_type(4)))  int    v4i;

#define EDIM 1024
#define KDIM 128
#define NBATCH 4
#define SEQ 2048

#ifndef __has_builtin
#define __has_builtin(x) 0
#endif

#if __has_builtin(__builtin_amdgcn_global_load_async_to_lds_b128)
#define HAVE_ASYNC_LDS 1
// builtin signature (from clang diagnostic): arg0 = int4 __device__* (AS1),
// arg1 = LDS destination (AS3), arg2/arg3 = imm offset / cpol.
#define ASYNC_GPTR(p) ((__attribute__((address_space(1))) v4i*)(p))
#define ASYNC_LPTR(p) ((__attribute__((address_space(3))) v4i*)(p))
#else
#define HAVE_ASYNC_LDS 0
#endif

static __device__ __forceinline__ void wait_async_lds() {
#if HAVE_ASYNC_LDS
#if __has_builtin(__builtin_amdgcn_s_wait_asynccnt)
    __builtin_amdgcn_s_wait_asynccnt(0);
#else
    asm volatile("s_wait_asynccnt 0" ::: "memory");
#endif
#endif
}

// ---- WMMA fragment loaders (layouts per CDNA5 ISA 7.12.2) -----------------
// A (16x32 bf16, row-major source): lane<16 -> M=lane, K = {k0..k0+7, k0+16..k0+23}
//                                   lane>=16 -> M=lane-16, K = {k0+8..15, k0+24..31}
static __device__ __forceinline__ v16bf load_a_frag(const bf16* __restrict__ A,
                                                    int lda, int m0, int k0, int lane) {
    const int half = lane >> 4, lm = lane & 15;
    const bf16* p = A + (size_t)(m0 + lm) * lda + k0 + half * 8;
    v8bf lo = *reinterpret_cast<const v8bf*>(p);
    v8bf hi = *reinterpret_cast<const v8bf*>(p + 16);
    v16bf r;
#pragma unroll
    for (int i = 0; i < 8; ++i) { r[i] = lo[i]; r[i + 8] = hi[i]; }
    return r;
}

// B (32x16 bf16) from N-major storage Bt[N][K]: lane<16 -> N=lane, K = {k0..k0+7, k0+16..k0+23}
//                                               lane>=16 -> N=lane-16, K = {k0+8..15, k0+24..31}
static __device__ __forceinline__ v16bf load_b_frag(const bf16* __restrict__ Bt,
                                                    int ldb, int n0, int k0, int lane) {
    const int half = lane >> 4, lm = lane & 15;
    const bf16* p = Bt + (size_t)(n0 + lm) * ldb + k0 + half * 8;
    v8bf lo = *reinterpret_cast<const v8bf*>(p);
    v8bf hi = *reinterpret_cast<const v8bf*>(p + 16);
    v16bf r;
#pragma unroll
    for (int i = 0; i < 8; ++i) { r[i] = lo[i]; r[i + 8] = hi[i]; }
    return r;
}

static __device__ __forceinline__ v8f wmma_bf16(v16bf a, v16bf b, v8f c) {
    return __builtin_amdgcn_wmma_f32_16x16x32_bf16(false, a, false, b, (short)0, c,
                                                   false, false);
}

// ---- fp32 -> bf16 conversion (vectorized x4, grid-stride) -----------------
__global__ __launch_bounds__(256) void k_cvt_bf16(const float* __restrict__ in,
                                                  bf16* __restrict__ out, long n) {
    long i = ((long)blockIdx.x * blockDim.x + threadIdx.x) * 4;
    long stride = (long)gridDim.x * blockDim.x * 4;
    for (; i + 3 < n; i += stride) {
        float4 f = *reinterpret_cast<const float4*>(in + i);
        v4bf o;
        o[0] = (bf16)f.x; o[1] = (bf16)f.y; o[2] = (bf16)f.z; o[3] = (bf16)f.w;
        *reinterpret_cast<v4bf*>(out + i) = o;
    }
}

// ---------------------------------------------------------------------------
// Tiled GEMM: out = A[M,K] x Bt[N,K]^T, per-batch (blockIdx.y).
// 256 threads compute a 128x128 block; K staged in 128x32 LDS tiles,
// double-buffered; waves register-block 4(M) x 2(N) WMMA tiles.
// mode 0: bf16 row-major (scaled)   mode 1: bf16 transposed (vT[b][f][s])
// mode 2: f32 row-major with per-batch offset
// ---------------------------------------------------------------------------
#define BM 128
#define BN 128
#define KT 32
#define LSTRIDE 40   // 32 + 8 pad: 80B rows, 16B aligned, conflict-free banks

static __device__ __forceinline__ void stage_tiles(const bf16* __restrict__ gA,
                                                   const bf16* __restrict__ gB,
                                                   int K, bf16* sA, bf16* sB,
                                                   int r, int c0, int k0) {
    const bf16* pa = gA + (size_t)r * K + k0 + c0;
    const bf16* pb = gB + (size_t)r * K + k0 + c0;
    bf16* qa = sA + r * LSTRIDE + c0;
    bf16* qb = sB + r * LSTRIDE + c0;
#if HAVE_ASYNC_LDS
    __builtin_amdgcn_global_load_async_to_lds_b128(ASYNC_GPTR(pa),     ASYNC_LPTR(qa),     0, 0);
    __builtin_amdgcn_global_load_async_to_lds_b128(ASYNC_GPTR(pa + 8), ASYNC_LPTR(qa + 8), 0, 0);
    __builtin_amdgcn_global_load_async_to_lds_b128(ASYNC_GPTR(pb),     ASYNC_LPTR(qb),     0, 0);
    __builtin_amdgcn_global_load_async_to_lds_b128(ASYNC_GPTR(pb + 8), ASYNC_LPTR(qb + 8), 0, 0);
#else
    *(v8bf*)qa       = *(const v8bf*)pa;
    *(v8bf*)(qa + 8) = *(const v8bf*)(pa + 8);
    *(v8bf*)qb       = *(const v8bf*)pb;
    *(v8bf*)(qb + 8) = *(const v8bf*)(pb + 8);
#endif
}

__global__ __launch_bounds__(256) void k_gemm_tiled(const bf16* __restrict__ A,
                                                    const bf16* __restrict__ Bt,
                                                    void* __restrict__ outv,
                                                    int M, int N, int K,
                                                    long batchStrideA, long batchStrideB,
                                                    float scale, int mode) {
    __shared__ bf16 lA[2][BM * LSTRIDE];
    __shared__ bf16 lB[2][BN * LSTRIDE];

    const int tid  = threadIdx.x;
    const int lane = tid & 31;
    const int wave = tid >> 5;
    const int wm = wave >> 2;            // 0..1 -> M-tiles wm*4 .. wm*4+3
    const int wn = wave & 3;             // 0..3 -> N-tiles wn*2 .. wn*2+1
    const int batch = blockIdx.y;
    const int nbn = N / BN;
    const int bm0 = (blockIdx.x / nbn) * BM;
    const int bn0 = (blockIdx.x % nbn) * BN;

    const bf16* gA = A  + (size_t)batch * batchStrideA + (size_t)bm0 * K;
    const bf16* gB = Bt + (size_t)batch * batchStrideB + (size_t)bn0 * K;

    const int r  = tid >> 1;             // 0..127: tile row staged by this thread
    const int c0 = (tid & 1) << 4;       // 0 or 16 (elements)

    stage_tiles(gA, gB, K, &lA[0][0], &lB[0][0], r, c0, 0);

    v8f acc[4][2] = {};
    const int nsteps = K / KT;
    for (int s = 0; s < nsteps; ++s) {
        const int cur = s & 1;
        wait_async_lds();                // staged data (this wave's issues) landed
        __syncthreads();                 // ... and everyone else's
        if (s + 1 < nsteps)
            stage_tiles(gA, gB, K, &lA[cur ^ 1][0], &lB[cur ^ 1][0], r, c0, (s + 1) * KT);

        v16bf af[4], bfr[2];
#pragma unroll
        for (int i = 0; i < 4; ++i)
            af[i] = load_a_frag(&lA[cur][0], LSTRIDE, (wm * 4 + i) * 16, 0, lane);
#pragma unroll
        for (int j = 0; j < 2; ++j)
            bfr[j] = load_b_frag(&lB[cur][0], LSTRIDE, (wn * 2 + j) * 16, 0, lane);
#pragma unroll
        for (int i = 0; i < 4; ++i)
#pragma unroll
            for (int j = 0; j < 2; ++j)
                acc[i][j] = wmma_bf16(af[i], bfr[j], acc[i][j]);
        __syncthreads();                 // done reading lA/lB[cur]
    }

    const int half = lane >> 4, lm = lane & 15;
#pragma unroll
    for (int i = 0; i < 4; ++i) {
        const int rbase = bm0 + (wm * 4 + i) * 16 + half * 8;
#pragma unroll
        for (int j = 0; j < 2; ++j) {
            const int col = bn0 + (wn * 2 + j) * 16 + lm;
#pragma unroll
            for (int rr = 0; rr < 8; ++rr) {
                const int grow = rbase + rr;
                if (mode == 0) {
                    ((bf16*)outv)[(size_t)grow * N + col] = (bf16)(acc[i][j][rr] * scale);
                } else if (mode == 1) {
                    const int bb = grow >> 11, ss = grow & (SEQ - 1);
                    ((bf16*)outv)[((size_t)bb * N + col) * SEQ + ss] = (bf16)acc[i][j][rr];
                } else {
                    ((float*)outv)[((size_t)batch * M + grow) * N + col] = acc[i][j][rr];
                }
            }
        }
    }
}

// ---- Fused scores + softmax: one block = 16 query rows x 2048 keys --------
// 8 waves; wave w owns key columns [w*256, w*256+256). Logits stay in VGPRs
// (16 accumulator tiles per wave). Row max/sum via shfl_xor + small LDS.
__global__ __launch_bounds__(256) void k_scores_softmax(const bf16* __restrict__ q,
                                                        const bf16* __restrict__ kmat,
                                                        bf16* __restrict__ probs) {
    const int b    = blockIdx.x >> 7;          // 128 query-blocks per batch
    const int q0   = (blockIdx.x & 127) << 4;  // query row base within batch
    const int wave = threadIdx.x >> 5;
    const int lane = threadIdx.x & 31;
    const int half = lane >> 4, lm = lane & 15;

    const bf16* qB = q    + (size_t)b * SEQ * KDIM;
    const bf16* kB = kmat + (size_t)b * SEQ * KDIM;

    v16bf afr[4];
#pragma unroll
    for (int i = 0; i < 4; ++i) afr[i] = load_a_frag(qB, KDIM, q0, i * 32, lane);

    v8f acc[16] = {};
#pragma unroll
    for (int t = 0; t < 16; ++t) {
        const int n0 = wave * 256 + t * 16;
#pragma unroll
        for (int i = 0; i < 4; ++i) {
            v16bf bfr = load_b_frag(kB, KDIM, n0, i * 32, lane);
            acc[t] = wmma_bf16(afr[i], bfr, acc[t]);
        }
    }

    __shared__ float sred[8][16];

    float gmax[8];
    {
        float rmax[8];
#pragma unroll
        for (int m = 0; m < 8; ++m) {
            float v = acc[0][m];
#pragma unroll
            for (int t = 1; t < 16; ++t) v = fmaxf(v, acc[t][m]);
#pragma unroll
            for (int off = 1; off < 16; off <<= 1) v = fmaxf(v, __shfl_xor(v, off, 32));
            rmax[m] = v;
        }
        if (lm == 0) {
#pragma unroll
            for (int m = 0; m < 8; ++m) sred[wave][half * 8 + m] = rmax[m];
        }
        __syncthreads();
#pragma unroll
        for (int m = 0; m < 8; ++m) {
            float v = sred[0][half * 8 + m];
#pragma unroll
            for (int wv = 1; wv < 8; ++wv) v = fmaxf(v, sred[wv][half * 8 + m]);
            gmax[m] = v;
        }
        __syncthreads();
    }

    float inv[8];
    {
        float rsum[8];
#pragma unroll
        for (int m = 0; m < 8; ++m) {
            float s = 0.f;
#pragma unroll
            for (int t = 0; t < 16; ++t) {
                float e = __expf(acc[t][m] - gmax[m]);
                acc[t][m] = e;
                s += e;
            }
#pragma unroll
            for (int off = 1; off < 16; off <<= 1) s += __shfl_xor(s, off, 32);
            rsum[m] = s;
        }
        if (lm == 0) {
#pragma unroll
            for (int m = 0; m < 8; ++m) sred[wave][half * 8 + m] = rsum[m];
        }
        __syncthreads();
#pragma unroll
        for (int m = 0; m < 8; ++m) {
            float s = 0.f;
#pragma unroll
            for (int wv = 0; wv < 8; ++wv) s += sred[wv][half * 8 + m];
            inv[m] = 1.0f / s;
        }
    }

    bf16* prow = probs + (size_t)b * SEQ * SEQ;
#pragma unroll
    for (int t = 0; t < 16; ++t) {
        const int col = wave * 256 + t * 16 + lm;
#pragma unroll
        for (int m = 0; m < 8; ++m) {
            const int row = q0 + half * 8 + m;
            prow[(size_t)row * SEQ + col] = (bf16)(acc[t][m] * inv[m]);
        }
    }
}

// ---------------------------------------------------------------------------
extern "C" void kernel_launch(void* const* d_in, const int* in_sizes, int n_in,
                              void* d_out, int out_size, void* d_ws, size_t ws_size,
                              hipStream_t stream) {
    (void)in_sizes; (void)n_in; (void)out_size; (void)ws_size;
    const float* x  = (const float*)d_in[0];   // [4,2048,1024]
    const float* Wq = (const float*)d_in[1];   // [128,1024]
    const float* Wk = (const float*)d_in[2];   // [128,1024]
    const float* Wv = (const float*)d_in[3];   // [1024,1024]
    float* out = (float*)d_out;                // [4,2048,1024] fp32

    char* ws = (char*)d_ws;
    size_t off = 0;
    bf16* xb  = (bf16*)(ws + off); off += (size_t)NBATCH * SEQ * EDIM * 2;   // 16 MB
    bf16* wqb = (bf16*)(ws + off); off += (size_t)KDIM * EDIM * 2;           // 256 KB
    bf16* wkb = (bf16*)(ws + off); off += (size_t)KDIM * EDIM * 2;           // 256 KB
    bf16* wvb = (bf16*)(ws + off); off += (size_t)EDIM * EDIM * 2;           // 2 MB
    bf16* qb  = (bf16*)(ws + off); off += (size_t)NBATCH * SEQ * KDIM * 2;   // 2 MB
    bf16* kb  = (bf16*)(ws + off); off += (size_t)NBATCH * SEQ * KDIM * 2;   // 2 MB
    bf16* vT  = (bf16*)(ws + off); off += (size_t)NBATCH * EDIM * SEQ * 2;   // 16 MB
    bf16* pr  = (bf16*)(ws + off); off += (size_t)NBATCH * SEQ * SEQ * 2;    // 32 MB

    const int M = NBATCH * SEQ;  // 8192 tokens
    const float qscale = 0.08838834764831845f;  // 1/sqrt(128)

    // 1) conversions
    auto cvt = [&](const float* in, bf16* o, long n) {
        long v = n / 4;
        int blocks = (int)((v + 255) / 256);
        if (blocks > 8192) blocks = 8192;
        k_cvt_bf16<<<blocks, 256, 0, stream>>>(in, o, n);
    };
    cvt(x,  xb,  (long)M * EDIM);
    cvt(Wq, wqb, (long)KDIM * EDIM);
    cvt(Wk, wkb, (long)KDIM * EDIM);
    cvt(Wv, wvb, (long)EDIM * EDIM);

    // 2) projections (tiled WMMA GEMM): q (scaled), k, v (stored transposed)
    k_gemm_tiled<<<dim3(M / BM, 1), 256, 0, stream>>>(
        xb, wqb, qb, M, KDIM, EDIM, 0, 0, qscale, 0);
    k_gemm_tiled<<<dim3(M / BM, 1), 256, 0, stream>>>(
        xb, wkb, kb, M, KDIM, EDIM, 0, 0, 1.0f, 0);
    k_gemm_tiled<<<dim3((M / BM) * (EDIM / BN), 1), 256, 0, stream>>>(
        xb, wvb, vT, M, EDIM, EDIM, 0, 0, 1.0f, 1);

    // 3) fused scores + softmax -> bf16 probabilities
    k_scores_softmax<<<NBATCH * (SEQ / 16), 256, 0, stream>>>(qb, kb, pr);

    // 4) P @ V -> fp32 output (batched tiled GEMM)
    k_gemm_tiled<<<dim3((SEQ / BM) * (EDIM / BN), NBATCH), 256, 0, stream>>>(
        pr, vT, out, SEQ, EDIM, SEQ,
        (long)SEQ * SEQ, (long)EDIM * SEQ, 1.0f, 2);
}
